// BottomUpIntegrator_43834436223550
// MI455X (gfx1250) — matrix operand order repaired and checked by reading, not compiled, not run on hardware
//
#include <hip/hip_runtime.h>
#include <math.h>

typedef __attribute__((ext_vector_type(16))) __bf16 v16bf;
typedef __attribute__((ext_vector_type(8)))  float  v8f;

// ---------------------------------------------------------------------------
// Kernel 1: per-cell MLP (36 -> 64 -> 1) via bf16 WMMA, fused sigmoid/weight.
// One wave = one 16-cell tile per iteration. W1 fragments resident in VGPRs.
// ---------------------------------------------------------------------------
__global__ __launch_bounds__(256) void cell_mlp_wmma(
    const float* __restrict__ state, const float* __restrict__ arch,
    const float* __restrict__ energy, const float* __restrict__ phil,
    const float* __restrict__ W1, const float* __restrict__ b1,
    const float* __restrict__ W2, const float* __restrict__ b2,
    float* __restrict__ wout, int N)
{
    const int lane = threadIdx.x & 31;
    const int hlf  = lane >> 4;     // which 16-lane half
    const int col  = lane & 15;     // A: M row; B/C: N column
    const int wavesPerBlock = blockDim.x >> 5;
    const int waveId = blockIdx.x * wavesPerBlock + (threadIdx.x >> 5);
    const int nWaves = gridDim.x * wavesPerBlock;
    const int tiles  = (N + 15) >> 4;

    // Resident B fragments of W1 (36x64, zero-padded to 64 in K).
    // B layout (32x16 bf16): VGPR v, lane l: n = 16*j + (l&15),
    //   K = (l>>4)*16 + 2v (+1 in high half of the dword).
    v16bf Bf[2][4];
#pragma unroll
    for (int g = 0; g < 2; ++g)
#pragma unroll
        for (int j = 0; j < 4; ++j)
#pragma unroll
            for (int v = 0; v < 8; ++v) {
                int k0 = hlf * 16 + 2 * v;
                int n  = j * 16 + col;
                int f0 = g * 32 + k0;
                float x0 = (f0     < 36) ? W1[f0 * 64 + n]       : 0.f;
                float x1 = (f0 + 1 < 36) ? W1[(f0 + 1) * 64 + n] : 0.f;
                Bf[g][j][2 * v]     = (__bf16)x0;
                Bf[g][j][2 * v + 1] = (__bf16)x1;
            }
    float b1j[4], w2j[4];
#pragma unroll
    for (int j = 0; j < 4; ++j) { b1j[j] = b1[j * 16 + col]; w2j[j] = W2[j * 16 + col]; }
    const float bias2 = b2[0];

    for (int tile = waveId; tile < tiles; tile += nWaves) {
        int cell = tile * 16 + col;
        int cc   = (cell < N) ? cell : 0;
        const float* srow = state + (size_t)cc * 32;
        const float* arow = arch  + (size_t)cc * 4;

        // A layout (16x32 bf16): lane holds M = l&15;
        // VGPR v: K = 16*(l>>4) + 2v (+1 in high half).
        v16bf A0, A1;
#pragma unroll
        for (int v = 0; v < 8; ++v) {
            int k0 = (hlf << 4) + (v << 1);
            float2 sv = *(const float2*)(srow + k0);
            A0[2 * v]     = (__bf16)sv.x;
            A0[2 * v + 1] = (__bf16)sv.y;
            float a0 = 0.f, a1 = 0.f;          // chunk1: features 32..35 = arch
            if (k0 == 0)      { a0 = arow[0]; a1 = arow[1]; }
            else if (k0 == 2) { a0 = arow[2]; a1 = arow[3]; }
            A1[2 * v]     = (__bf16)a0;
            A1[2 * v + 1] = (__bf16)a1;
        }

        v8f C[4] = {v8f{}, v8f{}, v8f{}, v8f{}};
#pragma unroll
        for (int j = 0; j < 4; ++j) {
            C[j] = __builtin_amdgcn_wmma_f32_16x16x32_bf16(
                false, A0, false, Bf[0][j], (short)0, C[j], false, false);
            C[j] = __builtin_amdgcn_wmma_f32_16x16x32_bf16(
                false, A1, false, Bf[1][j], (short)0, C[j], false, false);
        }

        // Layer 2: out[m] = sum_n relu(h[m,n]+b1[n]) * W2[n]
        // C layout: VGPR v, lane l -> (M = v + 8*(l>>4), n = 16*j + (l&15))
        float s[8];
#pragma unroll
        for (int v = 0; v < 8; ++v) {
            float acc = 0.f;
#pragma unroll
            for (int j = 0; j < 4; ++j) {
                float h = C[j][v] + b1j[j];
                h = fmaxf(h, 0.f);
                acc = fmaf(h, w2j[j], acc);
            }
            s[v] = acc;
        }
        // reduce over the 16 columns within each half
#pragma unroll
        for (int v = 0; v < 8; ++v) {
            s[v] += __shfl_xor(s[v], 1, 16);
            s[v] += __shfl_xor(s[v], 2, 16);
            s[v] += __shfl_xor(s[v], 4, 16);
            s[v] += __shfl_xor(s[v], 8, 16);
        }
        // redistribute: lane l obtains out[cell m = l&15]
        int sel = lane & 7;
        float r = s[0];
        if (sel == 1) r = s[1];
        if (sel == 2) r = s[2];
        if (sel == 3) r = s[3];
        if (sel == 4) r = s[4];
        if (sel == 5) r = s[5];
        if (sel == 6) r = s[6];
        if (sel == 7) r = s[7];
        int srcLane = ((col >> 3) << 4) | (col & 7);
        float outv = __shfl(r, srcLane, 32);

        float en = energy[cc], ph = phil[cc];
        float z    = outv + bias2;
        float base = 1.f / (1.f + __expf(-z));
        float ep   = en * ph;
        float imp  = fminf(fmaxf(base * ep, 0.01f), 1.f);
        float wv   = imp * ep;
        if (hlf == 0 && cell < N) wout[cell] = wv;
    }
}

// ---------------------------------------------------------------------------
// Kernel 2a: segment max of w. seg_ids sorted -> wave-segmented fmax scan,
// only the last lane of each segment-run issues the atomic (w>0, so int-bit
// compare == float compare).
// ---------------------------------------------------------------------------
__global__ __launch_bounds__(256) void seg_max_kernel(
    const float* __restrict__ w, const int* __restrict__ seg,
    int* __restrict__ segmax, int N)
{
    int i    = blockIdx.x * blockDim.x + threadIdx.x;
    int lane = threadIdx.x & 31;
    bool act = i < N;
    int   s  = act ? seg[i] : -1;
    float v  = act ? w[i]   : 0.f;
#pragma unroll
    for (int d = 1; d < 32; d <<= 1) {
        int   sU = __shfl_up(s, d, 32);
        float vU = __shfl_up(v, d, 32);
        if (lane >= d && sU == s) v = fmaxf(v, vU);
    }
    int sNext = __shfl_down(s, 1, 32);
    if (act && (lane == 31 || sNext != s))
        atomicMax(&segmax[s], __float_as_int(v));
}

// ---------------------------------------------------------------------------
// Kernel 2b: fused segment sums [e, e*arch(4), count, arch(4), arch^2(4),
// surprise]. Wave-segmented inclusive scan over sorted seg_ids; only the tail
// lane of each run emits atomics (~25-30x fewer L2 atomics).
// ---------------------------------------------------------------------------
__global__ __launch_bounds__(256) void seg_sum_kernel(
    const float* __restrict__ w, const int* __restrict__ seg,
    const float* __restrict__ arch, const float* __restrict__ surprise,
    const int* __restrict__ segmax, float* __restrict__ sums, int N)
{
    int i    = blockIdx.x * blockDim.x + threadIdx.x;
    int lane = threadIdx.x & 31;
    bool act = i < N;
    int  s   = act ? seg[i] : -1;

    float vals[15];
#pragma unroll
    for (int q = 0; q < 15; ++q) vals[q] = 0.f;
    if (act) {
        float m = __int_as_float(segmax[s]);
        float e = __expf(w[i] - m);
        float4 a = *(const float4*)(arch + (size_t)i * 4);
        vals[0]  = e;
        vals[1]  = e * a.x; vals[2]  = e * a.y;
        vals[3]  = e * a.z; vals[4]  = e * a.w;
        vals[5]  = 1.0f;
        vals[6]  = a.x;     vals[7]  = a.y;
        vals[8]  = a.z;     vals[9]  = a.w;
        vals[10] = a.x * a.x; vals[11] = a.y * a.y;
        vals[12] = a.z * a.z; vals[13] = a.w * a.w;
        vals[14] = surprise[i];
    }
    // segmented inclusive scan (Hillis-Steele with head-flag compare)
#pragma unroll
    for (int d = 1; d < 32; d <<= 1) {
        int sU = __shfl_up(s, d, 32);
        bool take = (lane >= d) && (sU == s);
#pragma unroll
        for (int q = 0; q < 15; ++q) {
            float u = __shfl_up(vals[q], d, 32);
            if (take) vals[q] += u;
        }
    }
    int sNext = __shfl_down(s, 1, 32);
    if (act && (lane == 31 || sNext != s)) {
        float* p = sums + (size_t)s * 16;
#pragma unroll
        for (int q = 0; q < 15; ++q) atomicAdd(p + q, vals[q]);
    }
}

// ---------------------------------------------------------------------------
// Kernel 3: per-cluster stats + tiny MLP (7 -> 32 -> 1); writes cluster_out rows
// ---------------------------------------------------------------------------
__global__ void cluster_kernel(const float* __restrict__ sums,
                               const float* __restrict__ V1, const float* __restrict__ c1,
                               const float* __restrict__ V2, const float* __restrict__ c2,
                               float* __restrict__ out,
                               float* __restrict__ impc_arr,
                               float* __restrict__ valid_arr, int K)
{
    int k = blockIdx.x * blockDim.x + threadIdx.x;
    if (k >= K) return;
    const float* p = sums + (size_t)k * 16;
    float esum = p[0];
    float cntN = p[5];
    bool valid = cntN > 0.f;

    float agg[4];
#pragma unroll
    for (int a = 0; a < 4; ++a) agg[a] = valid ? p[1 + a] / esum : 0.f;
    float mx = fmaxf(fmaxf(agg[0], agg[1]), fmaxf(agg[2], agg[3]));
    float ex[4], se = 0.f;
#pragma unroll
    for (int a = 0; a < 4; ++a) { ex[a] = __expf(agg[a] - mx); se += ex[a]; }
    float aggregate[4];
#pragma unroll
    for (int a = 0; a < 4; ++a) aggregate[a] = ex[a] / se;

    float cnt = fmaxf(cntN, 1.f);
    float varm = 0.f;
#pragma unroll
    for (int a = 0; a < 4; ++a) {
        float mean = p[6 + a] / cnt;
        float var  = (p[10 + a] - cntN * mean * mean) / fmaxf(cntN - 1.f, 1.f);
        varm += var;
    }
    varm *= 0.25f;
    float phi_c    = 1.f - fminf(1.f, varm * 2.f);
    float coh      = 1.f - varm;
    float pred_err = p[14] / cnt;
    float integ    = phi_c * (1.f - pred_err);

    float cf[7] = {aggregate[0], aggregate[1], aggregate[2], aggregate[3],
                   phi_c, coh, fminf(1.f, cntN / 20.f)};
    float o = c2[0];
    for (int h = 0; h < 32; ++h) {
        float hh = c1[h];
#pragma unroll
        for (int f = 0; f < 7; ++f) hh = fmaf(cf[f], V1[f * 32 + h], hh);
        hh = fmaxf(hh, 0.f);
        o = fmaf(hh, V2[h], o);
    }
    float basec = 1.f / (1.f + __expf(-o));
    float impc  = fminf(fmaxf(basec * phi_c, 0.01f), 1.f);

    float* row = out + (size_t)k * 8;
    row[0] = aggregate[0]; row[1] = aggregate[1];
    row[2] = aggregate[2]; row[3] = aggregate[3];
    row[4] = phi_c; row[5] = coh; row[6] = pred_err; row[7] = integ;

    impc_arr[k]  = valid ? impc : -__builtin_inff();
    valid_arr[k] = valid ? 1.f : 0.f;
}

// ---------------------------------------------------------------------------
// Kernel 4: single-block reduction over K clusters -> self_model[6]
// ---------------------------------------------------------------------------
__device__ __forceinline__ float blk_sum(float v, float* red) {
    int t = threadIdx.x;
    red[t] = v; __syncthreads();
    for (int s = 128; s > 0; s >>= 1) { if (t < s) red[t] += red[t + s]; __syncthreads(); }
    float r = red[0]; __syncthreads();
    return r;
}
__device__ __forceinline__ float blk_max(float v, float* red) {
    int t = threadIdx.x;
    red[t] = v; __syncthreads();
    for (int s = 128; s > 0; s >>= 1) { if (t < s) red[t] = fmaxf(red[t], red[t + s]); __syncthreads(); }
    float r = red[0]; __syncthreads();
    return r;
}
__device__ __forceinline__ int blk_or(int v, int* red) {
    int t = threadIdx.x;
    red[t] = v; __syncthreads();
    for (int s = 128; s > 0; s >>= 1) { if (t < s) red[t] |= red[t + s]; __syncthreads(); }
    int r = red[0]; __syncthreads();
    return r;
}

__global__ __launch_bounds__(256) void finalize_kernel(
    const float* __restrict__ rows, const float* __restrict__ impc_arr,
    const float* __restrict__ valid_arr, float* __restrict__ tail, int K)
{
    __shared__ float redf[256];
    __shared__ int   redi[256];
    int t = threadIdx.x;

    float mx = -__builtin_inff(), nv = 0.f, sphi = 0.f, scoh = 0.f;
    int mask = 0;
    for (int k = t; k < K; k += 256) {
        float ic = impc_arr[k];
        mx = fmaxf(mx, ic);
        float vf = valid_arr[k];
        nv += vf;
        const float* row = rows + (size_t)k * 8;
        sphi += row[4] * vf;
        scoh += row[5] * vf;
        if (vf > 0.f) {
            float b = row[0]; int bi = 0;
            if (row[1] > b) { b = row[1]; bi = 1; }
            if (row[2] > b) { b = row[2]; bi = 2; }
            if (row[3] > b) { b = row[3]; bi = 3; }
            mask |= 1 << bi;
        }
    }
    float maximp = blk_max(mx, redf);
    float nvT    = blk_sum(nv, redf);
    float sphiT  = blk_sum(sphi, redf);
    float scohT  = blk_sum(scoh, redf);
    int   maskT  = blk_or(mask, redi);

    float se = 0.f, g0 = 0.f, g1 = 0.f, g2 = 0.f, g3 = 0.f;
    for (int k = t; k < K; k += 256) {
        if (valid_arr[k] > 0.f) {
            float e = __expf(impc_arr[k] - maximp);
            se += e;
            const float* row = rows + (size_t)k * 8;
            g0 += e * row[0]; g1 += e * row[1];
            g2 += e * row[2]; g3 += e * row[3];
        }
    }
    float seT = blk_sum(se, redf);
    float g0T = blk_sum(g0, redf);
    float g1T = blk_sum(g1, redf);
    float g2T = blk_sum(g2, redf);
    float g3T = blk_sum(g3, redf);

    if (t == 0) {
        float n_valid = fmaxf(nvT, 1.f);
        float avg_phi = sphiT / n_valid;
        float vert    = scohT / n_valid;
        float unique  = (float)__popc(maskT);
        float phi_global = fminf(1.f, avg_phi * (0.5f + 0.5f * unique / 4.f));
        float ga[4] = {g0T / seT, g1T / seT, g2T / seT, g3T / seT};
        float m2 = fmaxf(fmaxf(ga[0], ga[1]), fmaxf(ga[2], ga[3]));
        float e0 = __expf(ga[0] - m2), e1 = __expf(ga[1] - m2);
        float e2 = __expf(ga[2] - m2), e3 = __expf(ga[3] - m2);
        float es = e0 + e1 + e2 + e3;
        tail[0] = e0 / es; tail[1] = e1 / es; tail[2] = e2 / es; tail[3] = e3 / es;
        tail[4] = phi_global; tail[5] = vert;
    }
}

// ---------------------------------------------------------------------------
extern "C" void kernel_launch(void* const* d_in, const int* in_sizes, int n_in,
                              void* d_out, int out_size, void* d_ws, size_t ws_size,
                              hipStream_t stream)
{
    const float* state    = (const float*)d_in[0];
    const float* arch     = (const float*)d_in[1];
    const float* energy   = (const float*)d_in[2];
    const float* phil     = (const float*)d_in[3];
    const float* surprise = (const float*)d_in[4];
    const int*   seg      = (const int*)  d_in[5];
    // d_in[6] = n_clusters scalar (derive K from out_size instead)
    const float* W1 = (const float*)d_in[7];
    const float* b1 = (const float*)d_in[8];
    const float* W2 = (const float*)d_in[9];
    const float* b2 = (const float*)d_in[10];
    const float* V1 = (const float*)d_in[11];
    const float* c1 = (const float*)d_in[12];
    const float* V2 = (const float*)d_in[13];
    const float* c2 = (const float*)d_in[14];

    const int N = in_sizes[2];             // energy is [N]
    const int K = (out_size - 6) / 8;      // cluster_out K*8 + self_model 6

    float* ws     = (float*)d_ws;
    float* wbuf   = ws;                          // [N]
    int*   segmax = (int*)(ws + (size_t)N);      // [K]
    float* sums   = ws + (size_t)N + K;          // [K*16]
    float* impc   = ws + (size_t)N + 17 * (size_t)K; // [K]
    float* validf = impc + K;                    // [K]

    float* out  = (float*)d_out;
    float* tail = out + (size_t)K * 8;

    // zero the atomic accumulators (segmax + sums are contiguous)
    hipMemsetAsync(segmax, 0, (size_t)(17 * K) * sizeof(float), stream);

    dim3 blk(256);
    cell_mlp_wmma<<<2048, blk, 0, stream>>>(state, arch, energy, phil,
                                            W1, b1, W2, b2, wbuf, N);
    seg_max_kernel<<<(N + 255) / 256, blk, 0, stream>>>(wbuf, seg, segmax, N);
    seg_sum_kernel<<<(N + 255) / 256, blk, 0, stream>>>(wbuf, seg, arch, surprise,
                                                        segmax, sums, N);
    cluster_kernel<<<(K + 255) / 256, blk, 0, stream>>>(sums, V1, c1, V2, c2,
                                                        out, impc, validf, K);
    finalize_kernel<<<1, blk, 0, stream>>>(out, impc, validf, tail, K);
}
